// MCNN_20839181320429
// MI455X (gfx1250) — matrix operand order repaired and checked by reading, not compiled, not run on hardware
//
#include <hip/hip_runtime.h>
#include <hip/hip_bf16.h>
#include <math.h>

typedef __attribute__((ext_vector_type(16))) __bf16 v16bf;
typedef __attribute__((ext_vector_type(8)))  float  v8f;
typedef __attribute__((ext_vector_type(4)))  int    v4i;

#define EPSBN 1e-5f

// ---- gfx1250 async VMEM->LDS path (guarded; falls back to sync copy) ----
#if defined(__has_builtin)
#  if __has_builtin(__builtin_amdgcn_global_load_async_to_lds_b128)
#    define HAS_ASYNC_LDS 1
#  endif
#endif
#ifndef HAS_ASYNC_LDS
#  define HAS_ASYNC_LDS 0
#endif

#if HAS_ASYNC_LDS
__device__ __forceinline__ void async_cp16(const void* g, void* l){
  __builtin_amdgcn_global_load_async_to_lds_b128(
      (__attribute__((address_space(1))) v4i*)(void*)(g),
      (__attribute__((address_space(3))) v4i*)(l), 0, 0);
}
__device__ __forceinline__ void async_wait0(){
#if __has_builtin(__builtin_amdgcn_s_wait_asynccnt)
  __builtin_amdgcn_s_wait_asynccnt(0);
#else
  asm volatile("s_wait_asynccnt 0" ::: "memory");
#endif
}
#endif

__device__ __forceinline__ float selu_f(float x){
  const float scale = 1.0507009873554805f;
  const float alpha = 1.6732632423543772f;
  return scale * (x > 0.f ? x : alpha * (expf(x) - 1.f));
}

// ---------------------------------------------------------------- workspace map (float slots)
static constexpr size_t AL(size_t x){ return (x + 63) & ~size_t(63); }
static constexpr size_t OF_POOLED1 = 0;                                   // f32 256*64*99 (accum)
static constexpr size_t OF_POOLED2 = OF_POOLED1 + AL(256*64*99);          // f32 256*64*100 (accum)
static constexpr size_t OF_STATS   = OF_POOLED2 + AL(256*64*100);         // f32 2048 (accum)
static constexpr size_t OF_MA      = OF_STATS   + 2048;                   // f32 256*9901
static constexpr size_t OF_PCA     = OF_MA      + AL(256*9901);           // f32 256*20
static constexpr size_t OF_CMEAN   = OF_PCA     + AL(256*20);             // f32 10000
static constexpr size_t OF_G       = OF_CMEAN   + AL(10000);              // f32 256*256
static constexpr size_t OF_U       = OF_G       + AL(65536);              // f32 256*20
static constexpr size_t OF_XCAT    = OF_U       + AL(256*20);             // f32 256*64*209
static constexpr size_t OF_H3      = OF_XCAT    + AL(256*64*209);         // f32 256*64*10
static constexpr size_t OF_HA      = OF_H3      + AL(256*64*10);          // f32 256*128*209
static constexpr size_t OF_FCIN    = OF_HA      + AL(256*128*209);        // f32 256*13312
static constexpr size_t OF_LGT     = OF_FCIN    + AL(256*13312);          // f32 256*256
static constexpr size_t OF_S1      = OF_LGT     + AL(65536);              // bf16 256*5000*32 (n,pos,c), reused
static constexpr size_t OF_XCBF    = OF_S1      + AL(20480000);           // bf16 256*10016 (padded stride)
static constexpr size_t OF_WABF    = OF_XCBF    + AL(1282048);            // bf16 64*1600 (o, k*32+c)
// stats sub-offsets (floats within OF_STATS)
static constexpr size_t ST_RAW = 0;     // 3 branches * 64ch * {sum,sumsq}
static constexpr size_t ST_BN1 = 384;   // 3 * 32 * 2
static constexpr size_t ST_PST = 576;   // 2 * 64 * 2
static constexpr size_t ST_AB  = 832;   // 3 * 64 * {alpha,beta}
static constexpr size_t ST_A1  = 1216;  // 64 * 2   (BNa1)
static constexpr size_t ST_A2  = 1344;  // 128 * 2  (BNa2)

// ---------------------------------------------------------------- small prep kernels
// reorder weights: wA2[o][k*32 + c] = w[o][c][k]  (im2col K = k*32+c)
__global__ __launch_bounds__(256) void k_cvt_w12(const float* __restrict__ w, __bf16* __restrict__ o2){
  int i = blockIdx.x*256 + threadIdx.x;
  if (i < 102400){
    int o = i / 1600, rem = i % 1600, k = rem >> 5, c = rem & 31;
    o2[i] = (__bf16)w[(o*32 + c)*50 + k];
  }
}

__global__ __launch_bounds__(256) void k_colmean(const float* __restrict__ x, float* __restrict__ cm){
  int j = blockIdx.x*256 + threadIdx.x;
  if (j >= 10000) return;
  float s = 0.f;
  for (int n = 0; n < 256; ++n) s += x[(long)n*10000 + j];
  cm[j] = s * (1.f/256.f);
}

// centered matrix, padded to stride 10016 with zeros -> guard-free WMMA staging
__global__ __launch_bounds__(256) void k_xc(const float* __restrict__ x, const float* __restrict__ cm,
                                            __bf16* __restrict__ xc){
  long i = (long)blockIdx.x*256 + threadIdx.x;
  if (i >= (long)256*10016) return;
  int j = (int)(i % 10016);
  long n = i / 10016;
  float v = (j < 10000) ? (x[n*10000 + j] - cm[j]) : 0.f;
  xc[i] = (__bf16)v;
}

// moving average (window 100) + min-max normalize per row
__global__ __launch_bounds__(256) void k_ma(const float* __restrict__ x, float* __restrict__ out){
  __shared__ float mas[9901];
  __shared__ float rmin[256], rmax[256];
  const int n = blockIdx.x, t = threadIdx.x;
  const float* row = x + (long)n*10000;
  float lmin = 3.4e38f, lmax = -3.4e38f;
  for (int i = t; i < 9901; i += 256){
    float s = 0.f;
    for (int k = 0; k < 100; ++k) s += row[i+k];
    s *= 0.01f;
    mas[i] = s;
    lmin = fminf(lmin, s); lmax = fmaxf(lmax, s);
  }
  rmin[t] = lmin; rmax[t] = lmax; __syncthreads();
  for (int s = 128; s > 0; s >>= 1){
    if (t < s){ rmin[t] = fminf(rmin[t], rmin[t+s]); rmax[t] = fmaxf(rmax[t], rmax[t+s]); }
    __syncthreads();
  }
  float mn = rmin[0], inv = 1.f/(rmax[0] - rmin[0]);
  for (int i = t; i < 9901; i += 256) out[(long)n*9901 + i] = (mas[i] - mn)*inv;
}

__global__ __launch_bounds__(256) void k_minmax_rows(const float* __restrict__ in, float* __restrict__ out, int L){
  __shared__ float rmin[256], rmax[256];
  const int n = blockIdx.x, t = threadIdx.x;
  float lmin = 3.4e38f, lmax = -3.4e38f;
  for (int i = t; i < L; i += 256){
    float v = in[(long)n*L + i];
    lmin = fminf(lmin, v); lmax = fmaxf(lmax, v);
  }
  rmin[t] = lmin; rmax[t] = lmax; __syncthreads();
  for (int s = 128; s > 0; s >>= 1){
    if (t < s){ rmin[t] = fminf(rmin[t], rmin[t+s]); rmax[t] = fmaxf(rmax[t], rmax[t+s]); }
    __syncthreads();
  }
  float mn = rmin[0], inv = 1.f/(rmax[0] - rmin[0]);
  for (int i = t; i < L; i += 256) out[(long)n*L + i] = (in[(long)n*L + i] - mn)*inv;
}

// ---------------------------------------------------------------- PCA: Gram via WMMA bf16
__global__ __launch_bounds__(256) void k_gram(const __bf16* __restrict__ Xc, float* __restrict__ G){
  const int bi = blockIdx.y, bj = blockIdx.x;
  const int t = threadIdx.x, wv = t >> 5, lane = t & 31, half = lane >> 4, l16 = lane & 15;
  const int rowt = wv & 3, colg = wv >> 2;
  __shared__ __align__(16) __bf16 Ab[64][32];
  __shared__ __align__(16) __bf16 Bb[64][32];
  v8f acc0 = {}; v8f acc1 = {};
  const int rid = t >> 1, hk = t & 1;
  const long grow = (rid < 64) ? (bi*64 + rid) : (bj*64 + (rid - 64));
  uint4* dst = (rid < 64) ? (uint4*)(&Ab[rid][hk*16]) : (uint4*)(&Bb[rid-64][hk*16]);
  union UB { v16bf v; uint4 q[2]; };
  for (int kc = 0; kc < 313; ++kc){
    __syncthreads();
    {
      const uint4* sp = (const uint4*)(Xc + grow*10016 + kc*32 + hk*16);
#if HAS_ASYNC_LDS
      async_cp16(sp + 0, dst + 0);
      async_cp16(sp + 1, dst + 1);
      async_wait0();
#else
      dst[0] = sp[0]; dst[1] = sp[1];
#endif
    }
    __syncthreads();
    UB a, b0, b1;
    a.q[0]  = *(const uint4*)(&Ab[rowt*16 + l16][half*16]);
    a.q[1]  = *(const uint4*)(&Ab[rowt*16 + l16][half*16 + 8]);
    b0.q[0] = *(const uint4*)(&Bb[(colg*2+0)*16 + l16][half*16]);
    b0.q[1] = *(const uint4*)(&Bb[(colg*2+0)*16 + l16][half*16 + 8]);
    b1.q[0] = *(const uint4*)(&Bb[(colg*2+1)*16 + l16][half*16]);
    b1.q[1] = *(const uint4*)(&Bb[(colg*2+1)*16 + l16][half*16 + 8]);
    acc0 = __builtin_amdgcn_wmma_f32_16x16x32_bf16(false, a.v, false, b0.v, (short)0, acc0, false, false);
    acc1 = __builtin_amdgcn_wmma_f32_16x16x32_bf16(false, a.v, false, b1.v, (short)0, acc1, false, false);
  }
  #pragma unroll
  for (int r = 0; r < 8; ++r){
    int gi = bi*64 + rowt*16 + r + 8*half;
    G[gi*256 + bj*64 + (colg*2+0)*16 + l16] = acc0[r];
    G[gi*256 + bj*64 + (colg*2+1)*16 + l16] = acc1[r];
  }
}

// subspace iteration for top-20 eigenvectors of G (256x256), one workgroup
__global__ __launch_bounds__(256) void k_eigen(const float* __restrict__ G, float* __restrict__ U){
  __shared__ float V[256][20], Vn[256][20], red[256], sc[20];
  __shared__ int ord[20];
  const int t = threadIdx.x;
  for (int q = 0; q < 20; ++q){
    unsigned h = (unsigned)(t*2654435761u) ^ (unsigned)(q*40503u + 12345u);
    V[t][q] = ((h >> 8) & 0xFFFF) * (1.f/65536.f) - 0.5f + (((t % 20) == q) ? 1.f : 0.f);
  }
  __syncthreads();
  for (int it = 0; it < 32; ++it){
    float acc[20];
    #pragma unroll
    for (int q = 0; q < 20; ++q) acc[q] = 0.f;
    for (int j = 0; j < 256; ++j){
      float g = G[t*256 + j];
      #pragma unroll
      for (int q = 0; q < 20; ++q) acc[q] += g * V[j][q];
    }
    #pragma unroll
    for (int q = 0; q < 20; ++q) Vn[t][q] = acc[q];
    __syncthreads();
    for (int q = 0; q < 20; ++q){
      for (int p = 0; p < q; ++p){
        red[t] = Vn[t][p] * Vn[t][q]; __syncthreads();
        for (int s = 128; s > 0; s >>= 1){ if (t < s) red[t] += red[t+s]; __syncthreads(); }
        float d = red[0]; __syncthreads();
        Vn[t][q] -= d * Vn[t][p]; __syncthreads();
      }
      red[t] = Vn[t][q] * Vn[t][q]; __syncthreads();
      for (int s = 128; s > 0; s >>= 1){ if (t < s) red[t] += red[t+s]; __syncthreads(); }
      float nrm = rsqrtf(fmaxf(red[0], 1e-30f)); __syncthreads();
      Vn[t][q] *= nrm; __syncthreads();
    }
    #pragma unroll
    for (int q = 0; q < 20; ++q) V[t][q] = Vn[t][q];
    __syncthreads();
  }
  for (int q = 0; q < 20; ++q){
    float w = 0.f;
    for (int j = 0; j < 256; ++j) w += G[t*256 + j] * V[j][q];
    red[t] = V[t][q] * w; __syncthreads();
    for (int s = 128; s > 0; s >>= 1){ if (t < s) red[t] += red[t+s]; __syncthreads(); }
    if (t == 0) sc[q] = red[0];
    __syncthreads();
  }
  if (t == 0){
    bool used[20] = {};
    for (int a = 0; a < 20; ++a){
      int best = 0; float bv = -3.4e38f;
      for (int q = 0; q < 20; ++q) if (!used[q] && sc[q] > bv){ bv = sc[q]; best = q; }
      used[best] = true; ord[a] = best;
    }
  }
  __syncthreads();
  for (int q = 0; q < 20; ++q) U[t*20 + q] = V[t][ord[q]];
}

// ---------------------------------------------------------------- stage 1
__global__ __launch_bounds__(256) void k_bn1_stats(const float* __restrict__ src, long total,
    const float* __restrict__ w1, const float* __restrict__ b1, float* __restrict__ out){
  const int c = blockIdx.y;
  const float w = w1[c], b = b1[c];
  float s = 0.f, q = 0.f;
  for (long i = (long)blockIdx.x*256 + threadIdx.x; i < total; i += (long)gridDim.x*256){
    float v = selu_f(w*src[i] + b);
    s += v; q += v*v;
  }
  __shared__ float red[256];
  int t = threadIdx.x;
  red[t] = s; __syncthreads();
  for (int k = 128; k > 0; k >>= 1){ if (t < k) red[t] += red[t+k]; __syncthreads(); }
  if (!t) atomicAdd(&out[c*2+0], red[0]);
  __syncthreads();
  red[t] = q; __syncthreads();
  for (int k = 128; k > 0; k >>= 1){ if (t < k) red[t] += red[t+k]; __syncthreads(); }
  if (!t) atomicAdd(&out[c*2+1], red[0]);
}

// writes transposed activations: s1t[n][j][c]  (c fastest -> coalesced, WMMA-friendly)
__global__ __launch_bounds__(256) void k_s1_write(const float* __restrict__ src,
    const float* __restrict__ w1, const float* __restrict__ b1,
    const float* __restrict__ g1, const float* __restrict__ be1,
    const float* __restrict__ st, int L, int Lout, __bf16* __restrict__ s1t){
  long i = (long)blockIdx.x*256 + threadIdx.x;
  long tot = (long)256*32*Lout;
  if (i >= tot) return;
  int c = (int)(i & 31);
  long rest = i >> 5;
  int j = (int)(rest % Lout);
  int n = (int)(rest / Lout);
  float cnt = 256.f * (float)L;
  float m = st[c*2+0] / cnt;
  float var = st[c*2+1] / cnt - m*m;
  float inv = rsqrtf(var + EPSBN);
  float w = w1[c], b = b1[c];
  const float* row = src + (long)n*L;
  float p = 0.5f*(selu_f(w*row[2*j] + b) + selu_f(w*row[2*j+1] + b));
  s1t[i] = (__bf16)((p - m)*inv*g1[c] + be1[c]);
}

// ---------------------------------------------------------------- k=50 conv via WMMA (hot loop)
// K-order = k*32+c; block computes 64ch x 128pos, 8 waves * 4 tiles; A straight from L2, B from LDS window.
__global__ __launch_bounds__(256) void k_conv12(const __bf16* __restrict__ s1t, const __bf16* __restrict__ wA,
    const float* __restrict__ bconv, float* __restrict__ pooled, float* __restrict__ stats, int Lin){
  const int n = blockIdx.y;
  const int l0 = blockIdx.x * 128;
  const int t = threadIdx.x, wv = t >> 5, lane = t & 31, half = lane >> 4, l16 = lane & 15;
  const int rowt = wv & 3, colg = wv >> 2;

  __shared__ __align__(16) __bf16 win2[177][32];   // positions l0-24 .. l0+152, (pos, c)
  __shared__ float lds_sum[64], lds_sq[64], lds_pool[64][4];

  for (int i = t; i < 64; i += 256){
    lds_sum[i] = 0.f; lds_sq[i] = 0.f;
    lds_pool[i][0] = 0.f; lds_pool[i][1] = 0.f; lds_pool[i][2] = 0.f; lds_pool[i][3] = 0.f;
  }
  const __bf16* srow = s1t + (long)n*Lin*32;
  if (l0 >= 24 && l0 + 153 <= Lin){
    const uint4* src = (const uint4*)(srow + (long)(l0 - 24)*32);
    uint4* dst = (uint4*)(&win2[0][0]);
#if HAS_ASYNC_LDS
    for (int i = t; i < 708; i += 256) async_cp16(src + i, dst + i);
    async_wait0();
#else
    for (int i = t; i < 708; i += 256) dst[i] = src[i];        // 177*32*2B / 16
#endif
  } else {
    for (int idx = t; idx < 177*32; idx += 256){
      int p = idx >> 5, c = idx & 31;
      int gp = l0 - 24 + p;
      float v = 0.f;
      if (gp >= 0 && gp < Lin) v = (float)srow[(long)gp*32 + c];
      win2[p][c] = (__bf16)v;
    }
  }
  __syncthreads();

  v8f acc0 = {}, acc1 = {}, acc2 = {}, acc3 = {};
  union UB { v16bf v; uint4 q[2]; };
  const __bf16* wrow = wA + (rowt*16 + l16)*1600;
  const int base = colg*64 + l16;
  for (int kc = 0; kc < 50; ++kc){
    UB a, b;
    a.q[0] = *(const uint4*)(wrow + kc*32 + half*16);
    a.q[1] = *(const uint4*)(wrow + kc*32 + half*16 + 8);
    b.q[0] = *(const uint4*)(&win2[base + kc +  0][half*16]);
    b.q[1] = *(const uint4*)(&win2[base + kc +  0][half*16 + 8]);
    acc0 = __builtin_amdgcn_wmma_f32_16x16x32_bf16(false, a.v, false, b.v, (short)0, acc0, false, false);
    b.q[0] = *(const uint4*)(&win2[base + kc + 16][half*16]);
    b.q[1] = *(const uint4*)(&win2[base + kc + 16][half*16 + 8]);
    acc1 = __builtin_amdgcn_wmma_f32_16x16x32_bf16(false, a.v, false, b.v, (short)0, acc1, false, false);
    b.q[0] = *(const uint4*)(&win2[base + kc + 32][half*16]);
    b.q[1] = *(const uint4*)(&win2[base + kc + 32][half*16 + 8]);
    acc2 = __builtin_amdgcn_wmma_f32_16x16x32_bf16(false, a.v, false, b.v, (short)0, acc2, false, false);
    b.q[0] = *(const uint4*)(&win2[base + kc + 48][half*16]);
    b.q[1] = *(const uint4*)(&win2[base + kc + 48][half*16 + 8]);
    acc3 = __builtin_amdgcn_wmma_f32_16x16x32_bf16(false, a.v, false, b.v, (short)0, acc3, false, false);
  }
  __syncthreads();
  const int bin0 = l0 / 50;
  #pragma unroll
  for (int r = 0; r < 8; ++r){
    int ch = rowt*16 + r + 8*half;
    float av[4] = {acc0[r], acc1[r], acc2[r], acc3[r]};
    #pragma unroll
    for (int tl = 0; tl < 4; ++tl){
      int l = l0 + colg*64 + tl*16 + l16;
      if (l < Lin){
        float v = selu_f(av[tl] + bconv[ch]);
        atomicAdd(&lds_sum[ch], v); atomicAdd(&lds_sq[ch], v*v);
        atomicAdd(&lds_pool[ch][l/50 - bin0], v);
      }
    }
  }
  __syncthreads();
  const int nbins = Lin / 50;
  for (int i = t; i < 64; i += 256){
    atomicAdd(&stats[i*2+0], lds_sum[i]);
    atomicAdd(&stats[i*2+1], lds_sq[i]);
  }
  for (int idx = t; idx < 64*4; idx += 256){
    int ch = idx >> 2, bb = idx & 3;
    int bin = bin0 + bb;
    if (bin < nbins) atomicAdd(&pooled[((long)n*64 + ch)*nbins + bin], lds_pool[ch][bb]);
  }
}

// branch 3: k=1 conv 32->64 over 10 positions + SELU + stats  (s1t is (n,pos,c))
__global__ __launch_bounds__(256) void k_branch3(const __bf16* __restrict__ s1t, const float* __restrict__ w3,
    const float* __restrict__ b3, float* __restrict__ h3, float* __restrict__ stats){
  const int n = blockIdx.x, t = threadIdx.x;
  __shared__ float xin[32][10];
  __shared__ float ss[64], sq[64];
  for (int i = t; i < 64; i += 256){ ss[i] = 0.f; sq[i] = 0.f; }
  for (int i = t; i < 320; i += 256){
    int l = i / 32, c = i % 32;
    xin[c][l] = (float)s1t[((long)n*10 + l)*32 + c];
  }
  __syncthreads();
  for (int idx = t; idx < 640; idx += 256){
    int o = idx / 10, l = idx % 10;
    float a = 0.f;
    for (int c = 0; c < 32; ++c) a += w3[o*32 + c] * xin[c][l];
    float v = selu_f(a + b3[o]);
    h3[((long)n*64 + o)*10 + l] = v;
    atomicAdd(&ss[o], v); atomicAdd(&sq[o], v*v);
  }
  __syncthreads();
  for (int i = t; i < 64; i += 256){
    atomicAdd(&stats[i*2+0], ss[i]);
    atomicAdd(&stats[i*2+1], sq[i]);
  }
}

__global__ __launch_bounds__(256) void k_poolstats(const float* __restrict__ pooled, int nbins,
                                                   float* __restrict__ pst){
  const int c = blockIdx.x, t = threadIdx.x;
  float s = 0.f, q = 0.f;
  for (int idx = t; idx < 256*nbins; idx += 256){
    int n = idx / nbins, bin = idx % nbins;
    float v = pooled[((long)n*64 + c)*nbins + bin] * 0.02f;
    s += v; q += v*v;
  }
  __shared__ float r0[256], r1[256];
  r0[t] = s; r1[t] = q; __syncthreads();
  for (int k = 128; k > 0; k >>= 1){ if (t < k){ r0[t] += r0[t+k]; r1[t] += r1[t+k]; } __syncthreads(); }
  if (!t){ pst[c*2+0] = r0[0]; pst[c*2+1] = r1[0]; }
}

// fold BN2 -> pool -> BN3 into per-channel alpha/beta (per branch)
__global__ void k_coef(const float* __restrict__ raw, const float* __restrict__ pst,
    const float* __restrict__ g2, const float* __restrict__ b2,
    const float* __restrict__ g3, const float* __restrict__ b3, float* __restrict__ ab){
  int t = threadIdx.x;
  if (t >= 192) return;
  int br = t / 64, c = t % 64;
  const float cntR[3] = {256.f*4950.f, 256.f*5000.f, 256.f*10.f};
  const float cntP[3] = {256.f*99.f,  256.f*100.f,  256.f*10.f};
  float m1 = raw[(br*64 + c)*2 + 0] / cntR[br];
  float v1 = raw[(br*64 + c)*2 + 1] / cntR[br] - m1*m1;
  float mp, vp;
  if (br < 2){
    mp = pst[(br*64 + c)*2 + 0] / cntP[br];
    vp = pst[(br*64 + c)*2 + 1] / cntP[br] - mp*mp;
  } else { mp = m1; vp = v1; }
  float inv1 = rsqrtf(v1 + EPSBN);
  float a2 = inv1 * g2[c];
  float mq = a2*(mp - m1) + b2[c];
  float vq = a2*a2*vp;
  float inv2 = rsqrtf(vq + EPSBN);
  ab[(br*64 + c)*2 + 0] = a2 * inv2 * g3[c];
  ab[(br*64 + c)*2 + 1] = (b2[c] - a2*m1 - mq)*inv2*g3[c] + b3[c];
}

// concat x1|x2|x3 -> x_cat (256,64,209) + BNa1 stats
__global__ __launch_bounds__(256) void k_concat(const float* __restrict__ p1, const float* __restrict__ p2,
    const float* __restrict__ h3, const float* __restrict__ ab,
    float* __restrict__ xcat, float* __restrict__ a1st){
  const int n = blockIdx.x, t = threadIdx.x;
  __shared__ float ss[64], sq[64];
  for (int i = t; i < 64; i += 256){ ss[i] = 0.f; sq[i] = 0.f; }
  __syncthreads();
  for (int idx = t; idx < 64*209; idx += 256){
    int c = idx / 209, pos = idx % 209;
    float p; int br;
    if (pos < 99){ p = p1[((long)n*64 + c)*99 + pos] * 0.02f; br = 0; }
    else if (pos < 199){ p = p2[((long)n*64 + c)*100 + (pos - 99)] * 0.02f; br = 1; }
    else { p = h3[((long)n*64 + c)*10 + (pos - 199)]; br = 2; }
    float v = ab[(br*64 + c)*2 + 0]*p + ab[(br*64 + c)*2 + 1];
    xcat[((long)n*64 + c)*209 + pos] = v;
    atomicAdd(&ss[c], v); atomicAdd(&sq[c], v*v);
  }
  __syncthreads();
  for (int i = t; i < 64; i += 256){
    atomicAdd(&a1st[i*2+0], ss[i]);
    atomicAdd(&a1st[i*2+1], sq[i]);
  }
}

// k=3 conv 64->128 with BNa1 folded into input, + SELU + BNa2 stats
__global__ __launch_bounds__(256) void k_conva(const float* __restrict__ xcat, const float* __restrict__ a1st,
    const float* __restrict__ ga1, const float* __restrict__ ba1,
    const float* __restrict__ wa, const float* __restrict__ bc,
    float* __restrict__ ha, float* __restrict__ a2st){
  const int n = blockIdx.x, t = threadIdx.x;
  __shared__ float xin[64][212];
  __shared__ float ss[128], sq[128];
  for (int i = t; i < 128; i += 256){ ss[i] = 0.f; sq[i] = 0.f; }
  for (int i = t; i < 64; i += 256){ xin[i][0] = 0.f; xin[i][210] = 0.f; }
  const float cnt = 256.f*209.f;
  for (int idx = t; idx < 64*209; idx += 256){
    int c = idx / 209, pos = idx % 209;
    float m = a1st[c*2+0] / cnt;
    float var = a1st[c*2+1] / cnt - m*m;
    float inv = rsqrtf(var + EPSBN);
    xin[c][pos+1] = (xcat[((long)n*64 + c)*209 + pos] - m)*inv*ga1[c] + ba1[c];
  }
  __syncthreads();
  for (int idx = t; idx < 128*209; idx += 256){
    int o = idx / 209, l = idx % 209;
    float a = 0.f;
    for (int c = 0; c < 64; ++c){
      const float* w = &wa[(o*64 + c)*3];
      a += w[0]*xin[c][l] + w[1]*xin[c][l+1] + w[2]*xin[c][l+2];
    }
    float v = selu_f(a + bc[o]);
    ha[((long)n*128 + o)*209 + l] = v;
    atomicAdd(&ss[o], v); atomicAdd(&sq[o], v*v);
  }
  __syncthreads();
  for (int i = t; i < 128; i += 256){
    atomicAdd(&a2st[i*2+0], ss[i]);
    atomicAdd(&a2st[i*2+1], sq[i]);
  }
}

__global__ __launch_bounds__(256) void k_pool2(const float* __restrict__ ha, const float* __restrict__ a2st,
    const float* __restrict__ ga2, const float* __restrict__ ba2, float* __restrict__ fcin){
  long i = (long)blockIdx.x*256 + threadIdx.x;
  if (i >= (long)256*128*104) return;
  int j = (int)(i % 104);
  int c = (int)((i / 104) % 128);
  int n = (int)(i / (104*128));
  const float cnt = 256.f*209.f;
  float m = a2st[c*2+0] / cnt;
  float var = a2st[c*2+1] / cnt - m*m;
  float inv = rsqrtf(var + EPSBN);
  float p = 0.5f*(ha[((long)n*128 + c)*209 + 2*j] + ha[((long)n*128 + c)*209 + 2*j + 1]);
  fcin[(long)n*13312 + c*104 + j] = (p - m)*inv*ga2[c] + ba2[c];
}

// fc1..fc4 fused per sample
__global__ __launch_bounds__(256) void k_fc(const float* __restrict__ fcin,
    const float* __restrict__ w1, const float* __restrict__ b1,
    const float* __restrict__ w2, const float* __restrict__ b2,
    const float* __restrict__ w3, const float* __restrict__ b3,
    const float* __restrict__ w4, const float* __restrict__ b4,
    float* __restrict__ logits){
  const int n = blockIdx.x, t = threadIdx.x;
  __shared__ float part[256*20];
  __shared__ float a1[20], a2[10], a3[20];
  float acc[20];
  #pragma unroll
  for (int q = 0; q < 20; ++q) acc[q] = 0.f;
  const float* xr = fcin + (long)n*13312;
  for (int i = t; i < 13312; i += 256){
    float xi = xr[i];
    #pragma unroll
    for (int q = 0; q < 20; ++q) acc[q] += xi * w1[q*13312 + i];
  }
  #pragma unroll
  for (int q = 0; q < 20; ++q) part[t*20 + q] = acc[q];
  __syncthreads();
  if (t < 20){
    float s = 0.f;
    for (int i = 0; i < 256; ++i) s += part[i*20 + t];
    a1[t] = selu_f(s + b1[t]);
  }
  __syncthreads();
  if (t < 10){
    float s = 0.f;
    for (int q = 0; q < 20; ++q) s += w2[t*20 + q]*a1[q];
    a2[t] = selu_f(s + b2[t]);
  }
  __syncthreads();
  if (t < 20){
    float s = 0.f;
    for (int r = 0; r < 10; ++r) s += w3[t*10 + r]*a2[r];
    a3[t] = selu_f(s + b3[t]);
  }
  __syncthreads();
  {
    float s = 0.f;
    for (int q = 0; q < 20; ++q) s += w4[t*20 + q]*a3[q];
    logits[(long)n*256 + t] = selu_f(s + b4[t]);
  }
}

// log_softmax over dim 0 (batch): one block per class column
__global__ __launch_bounds__(256) void k_lsm(const float* __restrict__ logits, float* __restrict__ out){
  const int j = blockIdx.x, t = threadIdx.x;
  __shared__ float red[256];
  float v = logits[(long)t*256 + j];
  red[t] = v; __syncthreads();
  for (int s = 128; s > 0; s >>= 1){ if (t < s) red[t] = fmaxf(red[t], red[t+s]); __syncthreads(); }
  float mx = red[0]; __syncthreads();
  red[t] = expf(v - mx); __syncthreads();
  for (int s = 128; s > 0; s >>= 1){ if (t < s) red[t] += red[t+s]; __syncthreads(); }
  float lse = mx + logf(red[0]);
  out[(long)t*256 + j] = v - lse;
}

// ---------------------------------------------------------------- host
extern "C" void kernel_launch(void* const* d_in, const int* in_sizes, int n_in,
                              void* d_out, int out_size, void* d_ws, size_t ws_size,
                              hipStream_t stream) {
  (void)in_sizes; (void)n_in; (void)out_size; (void)ws_size;
  const float* x       = (const float*)d_in[0];
  const float* w_c1    = (const float*)d_in[1];
  const float* b_c1    = (const float*)d_in[2];
  const float* g_bn1   = (const float*)d_in[3];
  const float* be_bn1  = (const float*)d_in[4];
  const float* w_c12   = (const float*)d_in[5];
  const float* b_c12   = (const float*)d_in[6];
  const float* w_c3    = (const float*)d_in[7];
  const float* b_c3    = (const float*)d_in[8];
  const float* g_bn2   = (const float*)d_in[9];
  const float* be_bn2  = (const float*)d_in[10];
  const float* g_bn3   = (const float*)d_in[11];
  const float* be_bn3  = (const float*)d_in[12];
  const float* g_a1    = (const float*)d_in[13];
  const float* be_a1   = (const float*)d_in[14];
  const float* w_ca    = (const float*)d_in[15];
  const float* b_ca    = (const float*)d_in[16];
  const float* g_a2    = (const float*)d_in[17];
  const float* be_a2   = (const float*)d_in[18];
  const float* w_fc1   = (const float*)d_in[19];
  const float* b_fc1   = (const float*)d_in[20];
  const float* w_fc2   = (const float*)d_in[21];
  const float* b_fc2   = (const float*)d_in[22];
  const float* w_fc3   = (const float*)d_in[23];
  const float* b_fc3   = (const float*)d_in[24];
  const float* w_fc4   = (const float*)d_in[25];
  const float* b_fc4   = (const float*)d_in[26];
  float* out = (float*)d_out;

  float*  ws    = (float*)d_ws;
  float*  pool1 = ws + OF_POOLED1;
  float*  pool2 = ws + OF_POOLED2;
  float*  st    = ws + OF_STATS;
  float*  ma    = ws + OF_MA;
  float*  pca   = ws + OF_PCA;
  float*  cmean = ws + OF_CMEAN;
  float*  G     = ws + OF_G;
  float*  U     = ws + OF_U;
  float*  xcat  = ws + OF_XCAT;
  float*  h3    = ws + OF_H3;
  float*  ha    = ws + OF_HA;
  float*  fcin  = ws + OF_FCIN;
  float*  lgt   = ws + OF_LGT;
  __bf16* s1t   = (__bf16*)(ws + OF_S1);
  __bf16* xcbf  = (__bf16*)(ws + OF_XCBF);
  __bf16* wabf  = (__bf16*)(ws + OF_WABF);

  // zero accumulators (pooled1|pooled2|stats contiguous)
  (void)hipMemsetAsync(pool1, 0, (AL(256*64*99) + AL(256*64*100) + 2048)*sizeof(float), stream);

  // weight reorder/convert for WMMA A-matrix
  k_cvt_w12<<<400, 256, 0, stream>>>(w_c12, wabf);

  // branch inputs
  k_ma<<<256, 256, 0, stream>>>(x, ma);
  k_colmean<<<40, 256, 0, stream>>>(x, cmean);
  k_xc<<<10016, 256, 0, stream>>>(x, cmean, xcbf);
  k_gram<<<dim3(4, 4), 256, 0, stream>>>(xcbf, G);
  k_eigen<<<1, 256, 0, stream>>>(G, U);
  k_minmax_rows<<<256, 256, 0, stream>>>(U, pca, 20);

  // branch 0 (moving-average): L=9901 -> s1 L=4950 -> conv12
  k_bn1_stats<<<dim3(64, 32), 256, 0, stream>>>(ma, (long)256*9901, w_c1, b_c1, st + ST_BN1 + 0);
  k_s1_write<<<(256*32*4950)/256, 256, 0, stream>>>(ma, w_c1, b_c1, g_bn1, be_bn1,
                                                    st + ST_BN1 + 0, 9901, 4950, s1t);
  k_conv12<<<dim3(39, 256), 256, 0, stream>>>(s1t, wabf, b_c12, pool1, st + ST_RAW + 0, 4950);

  // branch 1 (raw signal): L=10000 -> s1 L=5000 -> conv12
  k_bn1_stats<<<dim3(64, 32), 256, 0, stream>>>(x, (long)256*10000, w_c1, b_c1, st + ST_BN1 + 64);
  k_s1_write<<<(256*32*5000)/256, 256, 0, stream>>>(x, w_c1, b_c1, g_bn1, be_bn1,
                                                    st + ST_BN1 + 64, 10000, 5000, s1t);
  k_conv12<<<dim3(40, 256), 256, 0, stream>>>(s1t, wabf, b_c12, pool2, st + ST_RAW + 128, 5000);

  // branch 2 (PCA): L=20 -> s1 L=10 -> k=1 conv
  k_bn1_stats<<<dim3(4, 32), 256, 0, stream>>>(pca, (long)256*20, w_c1, b_c1, st + ST_BN1 + 128);
  k_s1_write<<<(256*32*10)/256, 256, 0, stream>>>(pca, w_c1, b_c1, g_bn1, be_bn1,
                                                  st + ST_BN1 + 128, 20, 10, s1t);
  k_branch3<<<256, 256, 0, stream>>>(s1t, w_c3, b_c3, h3, st + ST_RAW + 256);

  // BN2 -> pool -> BN3 folding
  k_poolstats<<<64, 256, 0, stream>>>(pool1, 99, st + ST_PST + 0);
  k_poolstats<<<64, 256, 0, stream>>>(pool2, 100, st + ST_PST + 128);
  k_coef<<<1, 192, 0, stream>>>(st + ST_RAW, st + ST_PST, g_bn2, be_bn2, g_bn3, be_bn3, st + ST_AB);

  // concat + BNa1 + conv(k=3) + BNa2 + pool2
  k_concat<<<256, 256, 0, stream>>>(pool1, pool2, h3, st + ST_AB, xcat, st + ST_A1);
  k_conva<<<256, 256, 0, stream>>>(xcat, st + ST_A1, g_a1, be_a1, w_ca, b_ca, ha, st + ST_A2);
  k_pool2<<<(256*128*104)/256, 256, 0, stream>>>(ha, st + ST_A2, g_a2, be_a2, fcin);

  // FC chain + log_softmax(dim=0)
  k_fc<<<256, 256, 0, stream>>>(fcin, w_fc1, b_fc1, w_fc2, b_fc2, w_fc3, b_fc3, w_fc4, b_fc4, lgt);
  k_lsm<<<256, 256, 0, stream>>>(lgt, out);
}
